// Block_17540646437178
// MI455X (gfx1250) — matrix operand order repaired and checked by reading, not compiled, not run on hardware
//
#include <hip/hip_runtime.h>
#include <math.h>

typedef __attribute__((ext_vector_type(16))) _Float16 v16h;
typedef __attribute__((ext_vector_type(8)))  _Float16 v8h;
typedef __attribute__((ext_vector_type(8)))  float    v8f;

#define NTOK  616
#define CDIM  768
#define HEADS 12
#define DHEAD 64
#define DFFN  3072
#define BATCH 16
#define TXTN  40
#define ROWS  (BATCH * NTOK)      /* 9856 */
#define QSCALE 0.125f             /* 64^-0.5 */

// ---------------------------------------------------------------------------
// helpers
// ---------------------------------------------------------------------------
__device__ __forceinline__ v16h ldpair(const _Float16* lo, const _Float16* hi) {
  union { v16h v; v8h h[2]; } u;
  u.h[0] = *(const v8h*)lo;
  u.h[1] = *(const v8h*)hi;
  return u.v;
}

__device__ __forceinline__ v8f wmma32(v16h a, v16h b, v8f c) {
  // D = A(16x32 f16) * B(32x16 f16) + C(16x16 f32)
  return __builtin_amdgcn_wmma_f32_16x16x32_f16(false, a, false, b, (short)0, c,
                                                false, false);
}

// CDNA5 async copy: global -> LDS, 16B per lane, tracked by ASYNCcnt.
__device__ __forceinline__ void async_copy_b128(void* lds_dst, const void* gsrc) {
  unsigned ldsoff = (unsigned)(size_t)lds_dst;   // generic ptr low 32b == LDS offset
  unsigned long long ga = (unsigned long long)(size_t)gsrc;
  asm volatile("global_load_async_to_lds_b128 %0, %1, off"
               :: "v"(ldsoff), "v"(ga) : "memory");
}
__device__ __forceinline__ void wait_async0() {
  asm volatile("s_wait_asynccnt 0" ::: "memory");
}

// compact row index -> global token row (text rows / image rows of each batch)
__device__ __forceinline__ int map_row(int r, int rm) {
  if (rm == 1) return (r / TXTN) * NTOK + (r % TXTN);
  if (rm == 2) return (r / (NTOK - TXTN)) * NTOK + TXTN + (r % (NTOK - TXTN));
  return r;
}

// ---------------------------------------------------------------------------
// fp32 -> f16 conversion (weights)
// ---------------------------------------------------------------------------
__global__ __launch_bounds__(256) void cvt_f32_f16(const float* __restrict__ s,
                                                   _Float16* __restrict__ d, int n) {
  int i = blockIdx.x * 256 + threadIdx.x;
  if (i < n) d[i] = (_Float16)s[i];
}

// ---------------------------------------------------------------------------
// LayerNorm (one row per block) -> f16 output. split!=0: pick params by token.
// ---------------------------------------------------------------------------
__global__ __launch_bounds__(256) void ln_f16_kernel(
    const float* __restrict__ x, const float* __restrict__ g0,
    const float* __restrict__ b0, const float* __restrict__ g1,
    const float* __restrict__ b1, _Float16* __restrict__ out, int split) {
  __shared__ float s1[256], s2[256];
  const int row = blockIdx.x, t = threadIdx.x;
  const float* xr = x + (size_t)row * CDIM;
  const int tok = row % NTOK;
  const float* G  = (split && tok >= TXTN) ? g1 : g0;
  const float* Bi = (split && tok >= TXTN) ? b1 : b0;
  float v0 = xr[t], v1 = xr[t + 256], v2 = xr[t + 512];
  s1[t] = v0 + v1 + v2;
  s2[t] = v0 * v0 + v1 * v1 + v2 * v2;
  __syncthreads();
  for (int o = 128; o > 0; o >>= 1) {
    if (t < o) { s1[t] += s1[t + o]; s2[t] += s2[t + o]; }
    __syncthreads();
  }
  const float mean = s1[0] * (1.0f / CDIM);
  const float var  = s2[0] * (1.0f / CDIM) - mean * mean;
  const float inv  = rsqrtf(var + 1e-5f);
  _Float16* orow = out + (size_t)row * CDIM;
  orow[t]       = (_Float16)((v0 - mean) * inv * G[t]       + Bi[t]);
  orow[t + 256] = (_Float16)((v1 - mean) * inv * G[t + 256] + Bi[t + 256]);
  orow[t + 512] = (_Float16)((v2 - mean) * inv * G[t + 512] + Bi[t + 512]);
}

// ---------------------------------------------------------------------------
// WMMA GEMM: C = A(M x K) * W^T + epilogue.
// Block: 256 thr = 8 waves in a 4(M) x 2(N) arrangement; block tile 128x128,
// each wave owns a 32x64 C tile (2 A-frags x 4 B-frags -> 8 WMMA / k-step,
// every B fragment reused by two WMMAs). The 128x32 weight tile is double-
// buffered in LDS via CDNA5 async global->LDS copies (ASYNCcnt) so the HBM/L2
// fetch of tile k+1 overlaps the matrix math of tile k.
// MODE 0: qkv scatter (+q scale)  MODE 1: proj + gamma1 residual -> f32
// MODE 2: fc1 + gelu -> f16       MODE 3: fc2 + gamma2 accumulate -> f32
// ---------------------------------------------------------------------------
template <int MODE>
__global__ __launch_bounds__(256) void gemm_wmma_kernel(
    const _Float16* __restrict__ A, const _Float16* __restrict__ W, int K,
    int rowmap, const float* __restrict__ bias_a, const float* __restrict__ bias_b,
    const float* __restrict__ resid, float* __restrict__ out32,
    _Float16* __restrict__ out16) {
  __shared__ _Float16 Bt[2][128 * 40];  // 128 weight rows x 32 k, pad stride 40
  const int lane = threadIdx.x & 31;
  const int wv   = threadIdx.x >> 5;
  const int g    = lane >> 4;
  const int ln   = lane & 15;
  const int nW   = (wv >> 2) * 64;                 // wave's 64-col half of 128
  const int n0   = blockIdx.x * 128 + nW;
  const int mB   = blockIdx.y * 128 + (wv & 3) * 32;
  const int arow0 = map_row(mB + ln, rowmap);
  const int arow1 = map_row(mB + 16 + ln, rowmap);
  const _Float16* aptr0 = A + (size_t)arow0 * K + g * 8;
  const _Float16* aptr1 = A + (size_t)arow1 * K + g * 8;

  // this thread's two 16B slices of the cooperative 128x32 weight-tile copy
  const int e0 = threadIdx.x * 2;          // transfer index (row = e>>2, ch = e&3)
  const int ti = e0 >> 2, ch = e0 & 3;
  const _Float16* wsrc = W + (size_t)(blockIdx.x * 128 + ti) * K + ch * 8;
  _Float16* wdst = &Bt[0][ti * 40 + ch * 8];
  const int bufstride = 128 * 40;

  // prologue: stage first tile asynchronously
  async_copy_b128(wdst, wsrc);
  async_copy_b128(wdst + 8, wsrc + 8);
  wait_async0();
  __syncthreads();

  v8f acc[2][4] = {};
  int p = 0;
  for (int k0 = 0; k0 < K; k0 += 32) {
    // kick off next weight tile while we do the math on the current one
    if (k0 + 32 < K) {
      async_copy_b128(wdst + (p ^ 1) * bufstride, wsrc + k0 + 32);
      async_copy_b128(wdst + (p ^ 1) * bufstride + 8, wsrc + k0 + 40);
    }
    __builtin_prefetch(aptr0 + k0 + 32, 0, 1);  // global_prefetch_b8 on A stream

    v16h a0 = ldpair(aptr0 + k0, aptr0 + k0 + 16);
    v16h a1 = ldpair(aptr1 + k0, aptr1 + k0 + 16);
    const _Float16* bbuf = &Bt[p][0];
#pragma unroll
    for (int ct = 0; ct < 4; ++ct) {
      const _Float16* bp = bbuf + (nW + ct * 16 + ln) * 40 + g * 8;
      v16h b = ldpair(bp, bp + 16);
      acc[0][ct] = wmma32(a0, b, acc[0][ct]);
      acc[1][ct] = wmma32(a1, b, acc[1][ct]);
    }
    wait_async0();     // own async writes to LDS landed
    __syncthreads();   // everyone's writes visible / reads of cur done
    p ^= 1;
  }

#pragma unroll
  for (int af = 0; af < 2; ++af) {
    const int mW = mB + af * 16;
#pragma unroll
    for (int ct = 0; ct < 4; ++ct) {
      const int j = n0 + ct * 16 + ln;
      if constexpr (MODE == 0) {
        const int mat = j / CDIM, jj = j % CDIM;
        const int hh = jj / DHEAD, d = jj % DHEAD;
        const float bias = (mat == 0) ? bias_a[jj] : (mat == 2 ? bias_b[jj] : 0.0f);
        const float sc = (mat == 0) ? QSCALE : 1.0f;
        _Float16* dst =
            out16 + (size_t)mat * ROWS * CDIM + (size_t)hh * NTOK * DHEAD + d;
#pragma unroll
        for (int v = 0; v < 8; ++v) {
          const int r = mW + v + 8 * g;
          const int bidx = r / NTOK, n = r % NTOK;
          const float val = (acc[af][ct][v] + bias) * sc;
          dst[(size_t)bidx * HEADS * NTOK * DHEAD + (size_t)n * DHEAD] =
              (_Float16)val;
        }
      } else if constexpr (MODE == 1) {
        const float bj = bias_a[j], gj = bias_b[j];
#pragma unroll
        for (int v = 0; v < 8; ++v) {
          const int r = mW + v + 8 * g;
          out32[(size_t)r * CDIM + j] =
              resid[(size_t)r * CDIM + j] + gj * (acc[af][ct][v] + bj);
        }
      } else if constexpr (MODE == 2) {
        const float bj = bias_a[j];
#pragma unroll
        for (int v = 0; v < 8; ++v) {
          const int gr = map_row(mW + v + 8 * g, rowmap);
          const float val = acc[af][ct][v] + bj;
          const float ge = 0.5f * val * (1.0f + erff(val * 0.70710678118654752f));
          out16[(size_t)gr * DFFN + j] = (_Float16)ge;
        }
      } else {
        const float bj = bias_a[j], gj = bias_b[j];
#pragma unroll
        for (int v = 0; v < 8; ++v) {
          const int gr = map_row(mW + v + 8 * g, rowmap);
          out32[(size_t)gr * CDIM + j] += gj * (acc[af][ct][v] + bj);
        }
      }
    }
  }
}

// ---------------------------------------------------------------------------
// Attention: one wave per (b, h, 16-query tile).
// S = Q K^T (WMMA) + rpb, mask -> softmax in LDS -> O = P V (WMMA).
// ---------------------------------------------------------------------------
__global__ __launch_bounds__(32) void attn_kernel(
    const _Float16* __restrict__ Q, const _Float16* __restrict__ Kx,
    const _Float16* __restrict__ V, const float* __restrict__ rpb,
    const int* __restrict__ mask, _Float16* __restrict__ O) {
  __shared__ float S[16 * 640];     // logits, then probabilities (f32)
  __shared__ _Float16 Vt[64 * 40];  // transposed V key-tile
  __shared__ float lsum[16];
  const int b = blockIdx.z, h = blockIdx.y, q0 = blockIdx.x * 16;
  const int lane = threadIdx.x, g = lane >> 4, ln = lane & 15;
  const size_t bh = (size_t)(b * HEADS + h) * NTOK;

  int qr = q0 + ln; if (qr > NTOK - 1) qr = NTOK - 1;
  const _Float16* qp = Q + (bh + qr) * DHEAD + g * 8;
  const v16h aq0 = ldpair(qp, qp + 16);        // d = 0..31
  const v16h aq1 = ldpair(qp + 32, qp + 48);   // d = 32..63
  const _Float16* kb = Kx + bh * DHEAD;
  const _Float16* vb = V + bh * DHEAD;
  const int* mk = mask + b * NTOK;

  // hoisted rpb row pointers for the 8 query rows this lane owns
  const float* rpr[8];
#pragma unroll
  for (int v = 0; v < 8; ++v) {
    int qa = q0 + v + 8 * g;
    int qc = qa < NTOK ? qa : NTOK - 1;
    rpr[v] = rpb + (size_t)h * NTOK * NTOK + (size_t)qc * NTOK;
  }

  // ---- pass 1: logits --------------------------------------------------
  for (int kt = 0; kt < 20; ++kt) {
    v8f c[2] = {};
#pragma unroll
    for (int s = 0; s < 2; ++s) {
#pragma unroll
      for (int ct = 0; ct < 2; ++ct) {
        int key = kt * 32 + ct * 16 + ln;
        int kc = key < NTOK ? key : NTOK - 1;
        const _Float16* kp = kb + (size_t)kc * DHEAD + s * 32 + g * 8;
        v16h bfr = ldpair(kp, kp + 16);
        c[ct] = wmma32(s ? aq1 : aq0, bfr, c[ct]);
      }
    }
#pragma unroll
    for (int ct = 0; ct < 2; ++ct) {
      const int col = kt * 32 + ct * 16 + ln;
      const int colc = col < NTOK ? col : NTOK - 1;
      const bool ok = (col < NTOK) && (mk[colc] != 0);
#pragma unroll
      for (int v = 0; v < 8; ++v) {
        const int row = v + 8 * g;
        const float sv = c[ct][v] + rpr[v][colc];
        S[row * 640 + col] = ok ? sv : -1e30f;
      }
    }
  }

  // ---- pass 2: softmax (exp written back in place) ---------------------
  for (int row = 0; row < 16; ++row) {
    float mv = -3.0e38f;
    for (int cc = lane; cc < 640; cc += 32) mv = fmaxf(mv, S[row * 640 + cc]);
#pragma unroll
    for (int o = 16; o > 0; o >>= 1) mv = fmaxf(mv, __shfl_xor(mv, o, 32));
    float sum = 0.0f;
    for (int cc = lane; cc < 640; cc += 32) {
      const float p = __expf(S[row * 640 + cc] - mv);
      S[row * 640 + cc] = p;
      sum += p;
    }
#pragma unroll
    for (int o = 16; o > 0; o >>= 1) sum += __shfl_xor(sum, o, 32);
    if (lane == 0) lsum[row] = sum;
  }

  // ---- pass 3: O = P V -------------------------------------------------
  v8f oacc[4] = {};
  for (int kt = 0; kt < 20; ++kt) {
    int key = kt * 32 + lane; if (key > NTOK - 1) key = NTOK - 1;
    const _Float16* vp = vb + (size_t)key * DHEAD;
#pragma unroll
    for (int d = 0; d < 64; d += 8) {
      v8h chunk = *(const v8h*)(vp + d);
#pragma unroll
      for (int j = 0; j < 8; ++j) Vt[(d + j) * 40 + lane] = chunk[j];
    }
    union { v16h v; _Float16 e[16]; } ua;
    const float* srow = &S[ln * 640 + kt * 32 + g * 8];
#pragma unroll
    for (int j = 0; j < 8; ++j) ua.e[j] = (_Float16)srow[j];
#pragma unroll
    for (int j = 0; j < 8; ++j) ua.e[8 + j] = (_Float16)srow[16 + j];
#pragma unroll
    for (int ct = 0; ct < 4; ++ct) {
      const _Float16* vt = &Vt[(ct * 16 + ln) * 40 + g * 8];
      v16h bfr = ldpair(vt, vt + 16);
      oacc[ct] = wmma32(ua.v, bfr, oacc[ct]);
    }
  }

#pragma unroll
  for (int ct = 0; ct < 4; ++ct) {
#pragma unroll
    for (int v = 0; v < 8; ++v) {
      const int row = v + 8 * g;
      const int qa = q0 + row;
      if (qa < NTOK) {
        const float val = oacc[ct][v] / lsum[row];
        O[((size_t)(b * NTOK + qa)) * CDIM + h * DHEAD + ct * 16 + ln] =
            (_Float16)val;
      }
    }
  }
}

// ---------------------------------------------------------------------------
// launch
// ---------------------------------------------------------------------------
extern "C" void kernel_launch(void* const* d_in, const int* in_sizes, int n_in,
                              void* d_out, int out_size, void* d_ws, size_t ws_size,
                              hipStream_t stream) {
  (void)in_sizes; (void)n_in; (void)out_size; (void)ws_size;
  const float* x      = (const float*)d_in[0];
  const int*   mask   = (const int*)d_in[1];
  const float* rpb    = (const float*)d_in[2];
  const float* w_qkv  = (const float*)d_in[3];
  const float* q_bias = (const float*)d_in[4];
  const float* v_bias = (const float*)d_in[5];
  const float* w_proj = (const float*)d_in[6];
  const float* b_proj = (const float*)d_in[7];
  const float* ln1_g  = (const float*)d_in[8];
  const float* ln1_b  = (const float*)d_in[9];
  const float* ln2t_g = (const float*)d_in[10];
  const float* ln2t_b = (const float*)d_in[11];
  const float* ln2i_g = (const float*)d_in[12];
  const float* ln2i_b = (const float*)d_in[13];
  const float* fc1t_w = (const float*)d_in[14];
  const float* fc1t_b = (const float*)d_in[15];
  const float* fc2t_w = (const float*)d_in[16];
  const float* fc2t_b = (const float*)d_in[17];
  const float* fc1i_w = (const float*)d_in[18];
  const float* fc1i_b = (const float*)d_in[19];
  const float* fc2i_w = (const float*)d_in[20];
  const float* fc2i_b = (const float*)d_in[21];
  const float* gamma1 = (const float*)d_in[22];
  const float* gamma2 = (const float*)d_in[23];
  float* out = (float*)d_out;

  _Float16* ws = (_Float16*)d_ws;
  size_t off = 0;
  _Float16* wqkv16  = ws + off; off += (size_t)3 * CDIM * CDIM;
  _Float16* wproj16 = ws + off; off += (size_t)CDIM * CDIM;
  _Float16* wf1t16  = ws + off; off += (size_t)DFFN * CDIM;
  _Float16* wf2t16  = ws + off; off += (size_t)CDIM * DFFN;
  _Float16* wf1i16  = ws + off; off += (size_t)DFFN * CDIM;
  _Float16* wf2i16  = ws + off; off += (size_t)CDIM * DFFN;
  _Float16* hbuf    = ws + off; off += (size_t)ROWS * CDIM;   // ln out / attn out / ln2 out
  _Float16* qkv16   = ws + off; off += (size_t)3 * ROWS * CDIM;
  _Float16* hff     = ws + off; off += (size_t)ROWS * DFFN;

  auto cvt = [&](const float* s, _Float16* d, int n) {
    cvt_f32_f16<<<(n + 255) / 256, 256, 0, stream>>>(s, d, n);
  };
  cvt(w_qkv, wqkv16, 3 * CDIM * CDIM);
  cvt(w_proj, wproj16, CDIM * CDIM);
  cvt(fc1t_w, wf1t16, DFFN * CDIM);
  cvt(fc2t_w, wf2t16, CDIM * DFFN);
  cvt(fc1i_w, wf1i16, DFFN * CDIM);
  cvt(fc2i_w, wf2i16, CDIM * DFFN);

  // LN1 -> h16
  ln_f16_kernel<<<ROWS, 256, 0, stream>>>(x, ln1_g, ln1_b, ln1_g, ln1_b, hbuf, 0);
  // QKV: 9856 x 768 x 2304  (N tiles of 128)
  gemm_wmma_kernel<0><<<dim3(18, 77), 256, 0, stream>>>(
      hbuf, wqkv16, CDIM, 0, q_bias, v_bias, nullptr, nullptr, qkv16);
  // attention -> o16 (reuses hbuf)
  attn_kernel<<<dim3(39, HEADS, BATCH), 32, 0, stream>>>(
      qkv16, qkv16 + (size_t)ROWS * CDIM, qkv16 + (size_t)2 * ROWS * CDIM,
      rpb, mask, hbuf);
  // proj + gamma1 residual -> d_out (f32)
  gemm_wmma_kernel<1><<<dim3(6, 77), 256, 0, stream>>>(
      hbuf, wproj16, CDIM, 0, b_proj, gamma1, x, out, nullptr);
  // LN2 (text/image params) -> h16
  ln_f16_kernel<<<ROWS, 256, 0, stream>>>(out, ln2t_g, ln2t_b, ln2i_g, ln2i_b,
                                          hbuf, 1);
  // FFN text rows (640 = 5*128) and image rows (9216 = 72*128)
  gemm_wmma_kernel<2><<<dim3(24, 5), 256, 0, stream>>>(
      hbuf, wf1t16, CDIM, 1, fc1t_b, nullptr, nullptr, nullptr, hff);
  gemm_wmma_kernel<2><<<dim3(24, 72), 256, 0, stream>>>(
      hbuf, wf1i16, CDIM, 2, fc1i_b, nullptr, nullptr, nullptr, hff);
  gemm_wmma_kernel<3><<<dim3(6, 5), 256, 0, stream>>>(
      hff, wf2t16, DFFN, 1, fc2t_b, gamma2, nullptr, out, nullptr);
  gemm_wmma_kernel<3><<<dim3(6, 72), 256, 0, stream>>>(
      hff, wf2i16, DFFN, 2, fc2i_b, gamma2, nullptr, out, nullptr);
}
